// ICTSP_197568496230
// MI455X (gfx1250) — matrix operand order repaired and checked by reading, not compiled, not run on hardware
//
#include <hip/hip_runtime.h>
#include <hip/hip_bf16.h>

typedef __attribute__((ext_vector_type(16))) _Float16 v16h;
typedef __attribute__((ext_vector_type(8)))  _Float16 v8h;
typedef __attribute__((ext_vector_type(8)))  float    v8f;

namespace {
constexpr int B_       = 64;
constexpr int N_       = 4096;
constexpr int D_       = 256;
constexpr int HALF_    = 128;
constexpr int T_       = 128;          // num_target_tokens
constexpr int O_       = N_ - T_;      // 3968 "other" tokens
constexpr int PD_      = 32;           // projection dim
constexpr int NUM_KEEP_= 396;          // int(O * 0.1)
constexpr int NFUSED_  = 398;          // 396 top + 2 group-fused
constexpr int OUT_T_   = NFUSED_ + T_; // 526 output tokens per batch
}

// ---------------------------------------------------------------------------
// Kernel 1: P = l2norm(x[:, token_offset:token_offset+rows, 128:256] @ W + b)
// One wave per 16-row tile. M=16, N=32, K=128 via 4x2 WMMA f32_16x16x32_f16.
// W staged TRANSPOSED in LDS so each B-fragment is one contiguous 32B LDS read.
// ---------------------------------------------------------------------------
__global__ void __launch_bounds__(256) proj_l2_kernel(
    const float* __restrict__ x, const float* __restrict__ W,
    const float* __restrict__ bias, _Float16* __restrict__ out,
    int token_offset, int rows)
{
  __shared__ __align__(32) _Float16 lwt[PD_ * HALF_];  // Wt[n][k], 32x128 f16, 8 KB
  const int tid = threadIdx.x;
  for (int i = tid; i < HALF_ * PD_; i += 256) {
    const int k = i >> 5, n = i & 31;                  // W is (128 x 32) row-major
    lwt[n * HALF_ + k] = (_Float16)W[i];
  }
  __syncthreads();

  const int wid  = tid >> 5, lane = tid & 31;
  const int tilesPerBatch = rows >> 4;
  const int gtile = blockIdx.x * 8 + wid;
  const int b    = gtile / tilesPerBatch;
  const int tile = gtile % tilesPerBatch;
  const int row0 = tile << 4;
  const int mrow = lane & 15;        // A: row M ; B: col N  (ISA frag layouts)
  const int kh   = lane >> 4;        // lane-half selects K sub-range
  const int kh8  = kh * 8;

  const float* xrow =
      x + ((size_t)b * N_ + token_offset + row0 + mrow) * D_ + HALF_;
  const _Float16* wt0 = lwt + (size_t)mrow * HALF_;         // column N = mrow
  const _Float16* wt1 = lwt + (size_t)(16 + mrow) * HALF_;  // column N = 16+mrow

  v8f c0 = {}; v8f c1 = {};
  #pragma unroll
  for (int kb = 0; kb < HALF_; kb += 32) {
    // A fragment 16x32 f16: lane holds K run [kh*16..kh*16+15] contiguous
    const float* s0 = xrow + kb;
    float4 f0 = *(const float4*)(s0 + kh * 16);
    float4 f1 = *(const float4*)(s0 + kh * 16 + 4);
    float4 f2 = *(const float4*)(s0 + kh * 16 + 8);
    float4 f3 = *(const float4*)(s0 + kh * 16 + 12);
    v16h a;
    a[0]=(_Float16)f0.x; a[1]=(_Float16)f0.y; a[2]=(_Float16)f0.z; a[3]=(_Float16)f0.w;
    a[4]=(_Float16)f1.x; a[5]=(_Float16)f1.y; a[6]=(_Float16)f1.z; a[7]=(_Float16)f1.w;
    a[8]=(_Float16)f2.x; a[9]=(_Float16)f2.y; a[10]=(_Float16)f2.z; a[11]=(_Float16)f2.w;
    a[12]=(_Float16)f3.x; a[13]=(_Float16)f3.y; a[14]=(_Float16)f3.z; a[15]=(_Float16)f3.w;

    // B fragments 32x16: lane = column N, K = kb + kh*16 + [0..15] contiguous
    const int krow0 = kb + kh * 16;
    v16h b0 = *(const v16h*)(wt0 + krow0);
    v16h b1 = *(const v16h*)(wt1 + krow0);

    c0 = __builtin_amdgcn_wmma_f32_16x16x32_f16(false, a, false, b0, (short)0, c0, false, false);
    c1 = __builtin_amdgcn_wmma_f32_16x16x32_f16(false, a, false, b1, (short)0, c1, false, false);
  }

  // bias (zeros in reference, kept for generality)
  const float bb0 = bias[mrow], bb1 = bias[16 + mrow];
  #pragma unroll
  for (int r = 0; r < 8; ++r) { c0[r] += bb0; c1[r] += bb1; }

  // L2-normalize each row. Row M = r + kh*8 lives in lanes of this 16-lane half.
  #pragma unroll
  for (int r = 0; r < 8; ++r) {
    float ss = c0[r]*c0[r] + c1[r]*c1[r];
    ss += __shfl_xor(ss, 1, 32);
    ss += __shfl_xor(ss, 2, 32);
    ss += __shfl_xor(ss, 4, 32);
    ss += __shfl_xor(ss, 8, 32);
    float inv = rsqrtf(fmaxf(ss, 1e-24f));
    c0[r] *= inv; c1[r] *= inv;
  }

  // Store f16 row-major (rows x 32)
  _Float16* orow = out + ((size_t)b * rows + row0) * PD_;
  #pragma unroll
  for (int r = 0; r < 8; ++r) {
    const int M = r + kh * 8;
    orow[(size_t)M * PD_ + mrow]      = (_Float16)c0[r];
    orow[(size_t)M * PD_ + 16 + mrow] = (_Float16)c1[r];
  }
}

// ---------------------------------------------------------------------------
// Kernel 2: scores[b,o] = (1/128) * sum_t relu(k_t . q_o)
// Block = (batch b, 128-o chunk); k staged in LDS; 8 WMMAs per wave.
// ---------------------------------------------------------------------------
__global__ void __launch_bounds__(256) sims_scores_kernel(
    const _Float16* __restrict__ kf, const _Float16* __restrict__ qf,
    float* __restrict__ scores)
{
  __shared__ __align__(16) _Float16 lk[T_ * PD_];   // 128x32 f16, 8 KB
  const int tid   = threadIdx.x;
  const int b     = blockIdx.x / (O_ / 128);
  const int chunk = blockIdx.x % (O_ / 128);

  const _Float16* kbase = kf + (size_t)b * T_ * PD_;
  for (int i = tid; i < (T_ * PD_) / 8; i += 256)
    ((v8h*)lk)[i] = ((const v8h*)kbase)[i];
  __syncthreads();

  const int wid = tid >> 5, lane = tid & 31;
  const int o0  = chunk * 128 + wid * 16;
  const int ncol = lane & 15, kh = lane >> 4;

  // B fragment: column N = o0+ncol of q^T, K = kh*16 + i (contiguous in q row)
  const _Float16* qsrc = qf + ((size_t)b * O_ + o0 + ncol) * PD_ + kh * 16;
  v16h bq = *(const v16h*)qsrc;

  float colsum = 0.f;
  #pragma unroll
  for (int tt = 0; tt < 8; ++tt) {
    const _Float16* s = lk + (size_t)(tt * 16 + ncol) * PD_ + kh * 16;
    v8h lo = *(const v8h*)s;
    v8h hi = *(const v8h*)(s + 8);
    v16h a;
    #pragma unroll
    for (int i = 0; i < 8; ++i) { a[i] = lo[i]; a[8 + i] = hi[i]; }
    v8f c = {};
    c = __builtin_amdgcn_wmma_f32_16x16x32_f16(false, a, false, bq, (short)0, c, false, false);
    #pragma unroll
    for (int r = 0; r < 8; ++r) colsum += fmaxf(c[r], 0.f);
  }
  colsum += __shfl_xor(colsum, 16, 32);   // combine the two row-halves
  if (lane < 16)
    scores[(size_t)b * O_ + o0 + lane] = colsum * (1.0f / (float)T_);
}

// ---------------------------------------------------------------------------
// Kernel 3: per-batch descending bitonic argsort (stable via index tie-break)
// ---------------------------------------------------------------------------
__global__ void __launch_bounds__(1024) bitonic_sort_kernel(
    const float* __restrict__ scores, float* __restrict__ sorted_scores,
    int* __restrict__ order)
{
  __shared__ float sv[4096];
  __shared__ int   si[4096];
  const int b = blockIdx.x, tid = threadIdx.x;
  for (int i = tid; i < 4096; i += 1024) {
    sv[i] = (i < O_) ? scores[(size_t)b * O_ + i] : -1e30f;
    si[i] = i;
  }
  __syncthreads();
  for (int k = 2; k <= 4096; k <<= 1) {
    for (int j = k >> 1; j > 0; j >>= 1) {
      for (int i = tid; i < 4096; i += 1024) {
        const int ixj = i ^ j;
        if (ixj > i) {
          const float a = sv[i], c = sv[ixj];
          const int ia = si[i], ic = si[ixj];
          const bool before  = (a > c) || (a == c && ia < ic); // desc, stable
          const bool descSeg = ((i & k) == 0);
          if (before != descSeg) {
            sv[i] = c; sv[ixj] = a; si[i] = ic; si[ixj] = ia;
          }
        }
      }
      __syncthreads();
    }
  }
  for (int i = tid; i < O_; i += 1024) {
    sorted_scores[(size_t)b * O_ + i] = sv[i];
    order[(size_t)b * O_ + i]         = si[i];
  }
}

// ---------------------------------------------------------------------------
// Kernel 4: the two softmax-fused group tokens (replicates reference quirk:
// sc gathered from sorted_scores at ORIGINAL token indices).
// Group 0: step=8, sorted positions [396..404) -> reversed out position 1
// Group 1: step=64, sorted positions [407..471) -> reversed out position 0
// ---------------------------------------------------------------------------
__global__ void __launch_bounds__(256) group_fuse_kernel(
    const float* __restrict__ x, const float* __restrict__ sorted_scores,
    const int* __restrict__ order, float* __restrict__ out)
{
  const int b = blockIdx.x >> 1, g = blockIdx.x & 1;
  const int members = g ? 64 : 8;
  const int pos0    = g ? 407 : 396;
  __shared__ float sc[64];
  __shared__ int   gi[64];
  const int tid = threadIdx.x;
  if (tid < members) {
    const int idx = order[(size_t)b * O_ + pos0 + tid];
    gi[tid] = idx;
    sc[tid] = sorted_scores[(size_t)b * O_ + idx];  // reference quirk
  }
  __syncthreads();

  float mx = -1e30f;
  for (int j = 0; j < members; ++j) mx = fmaxf(mx, sc[j]);
  float denom = 0.f;
  for (int j = 0; j < members; ++j) denom += expf(sc[j] - mx);
  const float inv = 1.f / denom;

  float acc = 0.f, gscore = 0.f;               // tid = feature d (0..255)
  for (int j = 0; j < members; ++j) {
    const float w = expf(sc[j] - mx) * inv;
    acc    += x[((size_t)b * N_ + gi[j]) * D_ + tid] * w;
    gscore += sc[j] * w;
  }
  const int p = 1 - g;                         // reversed position of f=396+g
  out[((size_t)b * OUT_T_ + p) * D_ + tid] = acc * gscore;
}

// ---------------------------------------------------------------------------
// Kernel 5: top-396 scaled tokens (reversed) + raw target tokens, float4-wide
// ---------------------------------------------------------------------------
__global__ void __launch_bounds__(64) scatter_kernel(
    const float* __restrict__ x, const float* __restrict__ sorted_scores,
    const int* __restrict__ order, float* __restrict__ out)
{
  const int y = blockIdx.x, b = blockIdx.y, d4 = threadIdx.x; // 64 * float4 = 256
  const float4* xv = (const float4*)x;
  float4* ov = (float4*)out;
  if (y < NUM_KEEP_) {
    const int   idx = order[(size_t)b * O_ + y];
    const float s   = sorted_scores[(size_t)b * O_ + y];
    float4 v = xv[((size_t)b * N_ + idx) * (D_ / 4) + d4];
    v.x *= s; v.y *= s; v.z *= s; v.w *= s;
    ov[((size_t)b * OUT_T_ + (NFUSED_ - 1 - y)) * (D_ / 4) + d4] = v;
  } else {
    const int t = y - NUM_KEEP_;
    ov[((size_t)b * OUT_T_ + NFUSED_ + t) * (D_ / 4) + d4] =
        xv[((size_t)b * N_ + O_ + t) * (D_ / 4) + d4];
  }
}

// ---------------------------------------------------------------------------
extern "C" void kernel_launch(void* const* d_in, const int* in_sizes, int n_in,
                              void* d_out, int out_size, void* d_ws, size_t ws_size,
                              hipStream_t stream) {
  (void)in_sizes; (void)n_in; (void)out_size; (void)ws_size;
  const float* x  = (const float*)d_in[0];
  const float* Wq = (const float*)d_in[1];
  const float* bq = (const float*)d_in[2];
  const float* Wk = (const float*)d_in[3];
  const float* bk = (const float*)d_in[4];
  float* out = (float*)d_out;

  char* p = (char*)d_ws;
  auto carve = [&](size_t bytes) -> char* {
    char* r = p; p += (bytes + 255) & ~(size_t)255; return r;
  };
  _Float16* qf     = (_Float16*)carve((size_t)B_ * O_ * PD_ * sizeof(_Float16));
  _Float16* kf     = (_Float16*)carve((size_t)B_ * T_ * PD_ * sizeof(_Float16));
  float*    scores = (float*)   carve((size_t)B_ * O_ * sizeof(float));
  float*    ssort  = (float*)   carve((size_t)B_ * O_ * sizeof(float));
  int*      order  = (int*)     carve((size_t)B_ * O_ * sizeof(int));

  proj_l2_kernel<<<(B_ * (O_ / 16)) / 8, 256, 0, stream>>>(x, Wq, bq, qf, 0, O_);
  proj_l2_kernel<<<(B_ * (T_ / 16)) / 8, 256, 0, stream>>>(x, Wk, bk, kf, O_, T_);
  sims_scores_kernel<<<B_ * (O_ / 128), 256, 0, stream>>>(kf, qf, scores);
  bitonic_sort_kernel<<<B_, 1024, 0, stream>>>(scores, ssort, order);
  group_fuse_kernel<<<B_ * 2, 256, 0, stream>>>(x, ssort, order, out);
  scatter_kernel<<<dim3(NUM_KEEP_ + T_, B_), 64, 0, stream>>>(x, ssort, order, out);
}
